// CABlock_CrossMamba_64802466562247
// MI455X (gfx1250) — compile-verified, matrix operand order checked by reading, not run on hardware
//
#include <hip/hip_runtime.h>
#include <hip/hip_bf16.h>
#include <math.h>

typedef _Float16 f16;
typedef _Float16 v16h __attribute__((ext_vector_type(16)));
typedef _Float16 v8h  __attribute__((ext_vector_type(8)));
typedef float    v8f  __attribute__((ext_vector_type(8)));

#define TOK     4096     // 8 * 512 tokens (all scales reshuffle the same 4096 tokens)
#define DIM     64
#define DI      128
#define NST     64
#define HID     128
#define SEQ     512
#define LN_EPS  1e-5f
#define BN_INV_C 0.99999500003749978f   // 1/sqrt(1+1e-5)
#define INV_SQRT512 0.044194173824159220f

// ---------------------------------------------------------------------------
// device helpers
// ---------------------------------------------------------------------------
__device__ __forceinline__ float silu_f(float x) { return x / (1.f + __expf(-x)); }
__device__ __forceinline__ float softplus_f(float x) {
  return (x > 20.f) ? x : log1pf(__expf(x));
}

// ---------------------------------------------------------------------------
// WMMA GEMM:  C[M][N] = A[M][K] * B[N][K]^T   (f16 in, f32 out)
// Wave computes a 16x64 output strip: one A fragment reused across 4 B
// fragments -> 4 v_wmma per k-step, 1/4 the A traffic.
// grid = (M/64, N/64), block = (32,4) (4 waves, each a different 16-row strip).
// Fragment lane layout per CDNA5 ISA 7.12.2 (16-bit A 16x32):
//   lane & 15   -> row (M for A, N for B)
//   (lane>>4)*8 -> K-base; halves [base..base+7] and [base+16..base+23]
// ---------------------------------------------------------------------------
__global__ void gemm_wmma_kernel(const f16* __restrict__ A, const f16* __restrict__ B,
                                 float* __restrict__ C, int K, int N) {
  const int m0   = (blockIdx.x * 4 + threadIdx.y) * 16;
  const int n0   = blockIdx.y * 64;
  const int lane = threadIdx.x;
  const int r16  = lane & 15;
  const int kb   = (lane >> 4) * 8;
  const f16* arow = A + (size_t)(m0 + r16) * K + kb;
  const f16* brow0 = B + (size_t)(n0 + r16) * K + kb;
  const f16* brow1 = brow0 + (size_t)16 * K;
  const f16* brow2 = brow0 + (size_t)32 * K;
  const f16* brow3 = brow0 + (size_t)48 * K;
  v8f acc0 = {}, acc1 = {}, acc2 = {}, acc3 = {};
  for (int k0 = 0; k0 < K; k0 += 32) {
    if (k0 + 32 < K) __builtin_prefetch(arow + k0 + 32, 0, 3);
    v8h alo = *(const v8h*)(arow + k0);
    v8h ahi = *(const v8h*)(arow + k0 + 16);
    v16h a;
#pragma unroll
    for (int i = 0; i < 8; ++i) { a[i] = alo[i]; a[8 + i] = ahi[i]; }
    {
      v8h blo = *(const v8h*)(brow0 + k0), bhi = *(const v8h*)(brow0 + k0 + 16);
      v16h b;
#pragma unroll
      for (int i = 0; i < 8; ++i) { b[i] = blo[i]; b[8 + i] = bhi[i]; }
      acc0 = __builtin_amdgcn_wmma_f32_16x16x32_f16(false, a, false, b, (short)0, acc0, false, false);
    }
    {
      v8h blo = *(const v8h*)(brow1 + k0), bhi = *(const v8h*)(brow1 + k0 + 16);
      v16h b;
#pragma unroll
      for (int i = 0; i < 8; ++i) { b[i] = blo[i]; b[8 + i] = bhi[i]; }
      acc1 = __builtin_amdgcn_wmma_f32_16x16x32_f16(false, a, false, b, (short)0, acc1, false, false);
    }
    {
      v8h blo = *(const v8h*)(brow2 + k0), bhi = *(const v8h*)(brow2 + k0 + 16);
      v16h b;
#pragma unroll
      for (int i = 0; i < 8; ++i) { b[i] = blo[i]; b[8 + i] = bhi[i]; }
      acc2 = __builtin_amdgcn_wmma_f32_16x16x32_f16(false, a, false, b, (short)0, acc2, false, false);
    }
    {
      v8h blo = *(const v8h*)(brow3 + k0), bhi = *(const v8h*)(brow3 + k0 + 16);
      v16h b;
#pragma unroll
      for (int i = 0; i < 8; ++i) { b[i] = blo[i]; b[8 + i] = bhi[i]; }
      acc3 = __builtin_amdgcn_wmma_f32_16x16x32_f16(false, a, false, b, (short)0, acc3, false, false);
    }
  }
  const int mb = m0 + (lane >> 4) * 8;
  float* crow = C + (size_t)mb * N + n0 + r16;
#pragma unroll
  for (int r = 0; r < 8; ++r) {
    crow[(size_t)r * N +  0] = acc0[r];
    crow[(size_t)r * N + 16] = acc1[r];
    crow[(size_t)r * N + 32] = acc2[r];
    crow[(size_t)r * N + 48] = acc3[r];
  }
}

// ---------------------------------------------------------------------------
// weight prep: f32 -> f16 copies (x_proj zero-padded 68->128 rows), A = -exp(A_log)
// ---------------------------------------------------------------------------
__global__ void prep_kernel(const float* __restrict__ inp,  const float* __restrict__ xp,
                            const float* __restrict__ cp,   const float* __restrict__ op,
                            const float* __restrict__ f1,   const float* __restrict__ f2,
                            const float* __restrict__ alog,
                            f16* wIn, f16* wXp, f16* wCp, f16* wOp, f16* wF1, f16* wF2,
                            float* Aneg) {
  const int i0 = blockIdx.x * blockDim.x + threadIdx.x;
  const int st = gridDim.x * blockDim.x;
  for (int i = i0; i < 256 * 64;  i += st) wIn[i] = (f16)inp[i];
  for (int i = i0; i < 128 * 128; i += st) wXp[i] = (i < 68 * 128) ? (f16)xp[i] : (f16)0.f;
  for (int i = i0; i < 64 * 64;   i += st) wCp[i] = (f16)cp[i];
  for (int i = i0; i < 64 * 128;  i += st) wOp[i] = (f16)op[i];
  for (int i = i0; i < 128 * 64;  i += st) wF1[i] = (f16)f1[i];
  for (int i = i0; i < 64 * 128;  i += st) wF2[i] = (f16)f2[i];
  for (int i = i0; i < 128 * 64;  i += st) Aneg[i] = -__expf(alog[i]);
}

// ---------------------------------------------------------------------------
// LayerNorm with multiscale gather permutation, f16 output.
// One wave per token (wave32: lane handles ch and ch+32).
// T==512 makes the permutation the identity.
// ---------------------------------------------------------------------------
__global__ void ln_gather_f16_kernel(const float* __restrict__ x,
                                     const float* __restrict__ g, const float* __restrict__ b,
                                     f16* __restrict__ out, int T) {
  const int wid  = threadIdx.x >> 5;
  const int lane = threadIdx.x & 31;
  const int tok  = blockIdx.x * (blockDim.x >> 5) + wid;
  const int sb = tok / T, t = tok - sb * T;
  const int sidx = sb >> 3, bb = sb & 7;
  const int orig = bb * SEQ + sidx * T + t;
  const float* xr = x + (size_t)orig * DIM;
  float a0 = xr[lane], a1 = xr[lane + 32];
  float s = a0 + a1, q = a0 * a0 + a1 * a1;
#pragma unroll
  for (int off = 16; off >= 1; off >>= 1) { s += __shfl_xor(s, off); q += __shfl_xor(q, off); }
  const float mean = s * (1.f / 64.f);
  const float var  = q * (1.f / 64.f) - mean * mean;
  const float rstd = rsqrtf(var + LN_EPS);
  f16* o = out + (size_t)tok * DIM;
  o[lane]      = (f16)((a0 - mean) * rstd * g[lane]      + b[lane]);
  o[lane + 32] = (f16)((a1 - mean) * rstd * g[lane + 32] + b[lane + 32]);
}

// x2 = x + LN(xo)  (natural token order, f32 out)
__global__ void ln_add_kernel(const float* __restrict__ xo, const float* __restrict__ x,
                              const float* __restrict__ g, const float* __restrict__ b,
                              float* __restrict__ x2) {
  const int wid  = threadIdx.x >> 5;
  const int lane = threadIdx.x & 31;
  const int tok  = blockIdx.x * (blockDim.x >> 5) + wid;
  const float* r = xo + (size_t)tok * DIM;
  float a0 = r[lane], a1 = r[lane + 32];
  float s = a0 + a1, q = a0 * a0 + a1 * a1;
#pragma unroll
  for (int off = 16; off >= 1; off >>= 1) { s += __shfl_xor(s, off); q += __shfl_xor(q, off); }
  const float mean = s * (1.f / 64.f);
  const float var  = q * (1.f / 64.f) - mean * mean;
  const float rstd = rsqrtf(var + LN_EPS);
  const float* xr = x + (size_t)tok * DIM;
  float* o = x2 + (size_t)tok * DIM;
  o[lane]      = xr[lane]      + (a0 - mean) * rstd * g[lane]      + b[lane];
  o[lane + 32] = xr[lane + 32] + (a1 - mean) * rstd * g[lane + 32] + b[lane + 32];
}

// ---------------------------------------------------------------------------
// depthwise causal conv (width 4, left pad 3) + bias + SiLU
// uz: [tok][256] (u = cols 0..127); out: us f32 + f16 copies [tok][128]
// ---------------------------------------------------------------------------
__global__ void conv_silu_kernel(const float* __restrict__ uz, const float* __restrict__ w,
                                 const float* __restrict__ wb, float* __restrict__ us,
                                 f16* __restrict__ us16, int T) {
  const int idx = blockIdx.x * blockDim.x + threadIdx.x;
  if (idx >= TOK * DI) return;
  const int tok = idx >> 7, d = idx & 127;
  const int t = tok % T;                    // position within this sub-sequence
  float acc = wb[d];
  const float* w4 = w + d * 4;
#pragma unroll
  for (int j = 0; j < 4; ++j) {
    const int tt = t - 3 + j;
    if (tt >= 0) acc += w4[j] * uz[(size_t)(tok - 3 + j) * 256 + d];
  }
  const float v = silu_f(acc);
  us[idx]   = v;
  us16[idx] = (f16)v;
}

// dt = softplus(dtr @ dt_w.T + dt_b);  dtr = dbl[:, 0:4]  (dbl row stride 128)
__global__ void dt_softplus_kernel(const float* __restrict__ dbl, const float* __restrict__ dtw,
                                   const float* __restrict__ dtb, float* __restrict__ dto) {
  const int idx = blockIdx.x * blockDim.x + threadIdx.x;
  if (idx >= TOK * DI) return;
  const int tok = idx >> 7, d = idx & 127;
  const float* dr = dbl + (size_t)tok * 128;
  float a = dtb[d];
#pragma unroll
  for (int r = 0; r < 4; ++r) a += dr[r] * dtw[d * 4 + r];
  dto[idx] = softplus_f(a);
}

__global__ void cast_f16_kernel(const float* __restrict__ in, f16* __restrict__ out, int n) {
  const int i = blockIdx.x * blockDim.x + threadIdx.x;
  if (i < n) out[i] = (f16)in[i];
}

// ---------------------------------------------------------------------------
// selective scan: block per sub-sequence, thread per channel d.
// h[64] state + A-row in VGPRs. Per-timestep B_t/C_t broadcast is a
// double-buffered CDNA5 async memory->LDS pipeline: global_load_async_to_lds
// for t+1 is issued before computing t, hidden behind 64 exp+FMA chains.
// bc[buf][0..63] = B_t, bc[buf][64..127] = C_t.
// ---------------------------------------------------------------------------
__global__ void __launch_bounds__(128) scan_kernel(
    const float* __restrict__ us, const float* __restrict__ dt,
    const float* __restrict__ Aneg, const float* __restrict__ dbl,
    const float* __restrict__ Cm, const float* __restrict__ Dp,
    float* __restrict__ y, int T) {
  const int sb = blockIdx.x;
  const int d  = threadIdx.x;
  float h[NST], Ar[NST];
#pragma unroll
  for (int s = 0; s < NST; ++s) { h[s] = 0.f; Ar[s] = Aneg[d * NST + s]; }
  const float Dv = Dp[d];
  __shared__ float bc[2][128];
  // per-lane source stream (lane d<64 streams B, else C) and LDS targets
  const float* src = (d < 64) ? (dbl + (size_t)sb * T * 128 + 4 + d)
                              : (Cm  + (size_t)sb * T * 64 + (d - 64));
  const size_t sstride = (d < 64) ? 128 : 64;
  const unsigned lds0 = (unsigned)(size_t)&bc[0][d];   // low 32 bits of LDS aperture = LDS offset
  const unsigned lds1 = (unsigned)(size_t)&bc[1][d];
  // prologue: async load for t = 0 into buffer 0
  asm volatile("global_load_async_to_lds_b32 %0, %1, off"
               : : "v"(lds0), "v"(src) : "memory");
  for (int t = 0; t < T; ++t) {
    asm volatile("s_wait_asynccnt 0x0" ::: "memory");
    __syncthreads();
    if (t + 1 < T) {
      const float* nsrc = src + (size_t)(t + 1) * sstride;
      const unsigned nlds = ((t + 1) & 1) ? lds1 : lds0;
      asm volatile("global_load_async_to_lds_b32 %0, %1, off"
                   : : "v"(nlds), "v"(nsrc) : "memory");
    }
    const float* Bs = bc[t & 1];
    const float* Cs = bc[t & 1] + 64;
    const int tok = sb * T + t;
    const float dtv = dt[(size_t)tok * DI + d];
    const float uv  = us[(size_t)tok * DI + d];
    const float dbu = dtv * uv;
    float acc = 0.f;
#pragma unroll
    for (int s = 0; s < NST; ++s) {
      const float dA = __expf(dtv * Ar[s]);
      h[s] = dA * h[s] + dbu * Bs[s];
      acc += h[s] * Cs[s];
    }
    y[(size_t)tok * DI + d] = acc + uv * Dv;
    __syncthreads();   // all lanes done reading bc[t&1] before t+1 issues into bc[(t+2)&1]
  }
}

// yg = f16( y * silu(z) ),  z = uz[:,128:256]
__global__ void gate_kernel(const float* __restrict__ y, const float* __restrict__ uz,
                            f16* __restrict__ yg) {
  const int idx = blockIdx.x * blockDim.x + threadIdx.x;
  if (idx >= TOK * DI) return;
  const int tok = idx >> 7, d = idx & 127;
  const float z = uz[(size_t)tok * 256 + 128 + d];
  yg[idx] = (f16)(y[idx] * silu_f(z));
}

// scatter ca_layer output back to original token order, accumulating into x_o
__global__ void scatter_add_kernel(const float* __restrict__ yo, float* __restrict__ xo,
                                   int T, int beta) {
  const int idx = blockIdx.x * blockDim.x + threadIdx.x;
  if (idx >= TOK * DIM) return;
  const int tok = idx >> 6, ch = idx & 63;
  const int sb = tok / T, t = tok - sb * T;
  const int sidx = sb >> 3, bb = sb & 7;
  const int orig = bb * SEQ + sidx * T + t;
  float* dst = xo + (size_t)orig * DIM + ch;
  *dst = (beta ? *dst : 0.f) + yo[idx];
}

__global__ void bn_relu_kernel(float* __restrict__ h, const float* __restrict__ g,
                               const float* __restrict__ b) {
  const int idx = blockIdx.x * blockDim.x + threadIdx.x;
  if (idx >= TOK * HID) return;
  const int ch = idx & 127;
  h[idx] = fmaxf(h[idx] * BN_INV_C * g[ch] + b[ch], 0.f);
}

// ---------------------------------------------------------------------------
// per-(batch,channel) 512-pt FFT -> diag spectral ReLU filter -> IFFT (real).
// DIF forward (natural->bitrev) + DIT inverse (bitrev->natural): the filter is
// uniform over frequency, so no bit-reversal pass is needed at all.
// ---------------------------------------------------------------------------
__global__ void __launch_bounds__(256) fft_filter_kernel(
    float* __restrict__ h, f16* __restrict__ h16,
    const float* __restrict__ r, const float* __restrict__ ii,
    const float* __restrict__ rb, const float* __restrict__ ib) {
  const int b   = blockIdx.x;
  const int ch  = blockIdx.y;
  const int tid = threadIdx.x;
  __shared__ float re[SEQ], im[SEQ];
  float* base = h + (size_t)b * SEQ * HID + ch;
  re[tid] = base[(size_t)tid * HID];
  re[tid + 256] = base[(size_t)(tid + 256) * HID];
  im[tid] = 0.f; im[tid + 256] = 0.f;
  __syncthreads();
  // forward DIF (e^{-i})
  for (int len = SEQ; len >= 2; len >>= 1) {
    const int half = len >> 1;
    const int grp = tid / half, pos = tid - grp * half;
    const int i0 = grp * len + pos, i1 = i0 + half;
    const float ang = -6.28318530717958647692f * (float)pos / (float)len;
    const float wr = __cosf(ang), wi = __sinf(ang);
    const float ur = re[i0], ui = im[i0], vr = re[i1], vi = im[i1];
    const float dr = ur - vr, di = ui - vi;
    re[i0] = ur + vr; im[i0] = ui + vi;
    re[i1] = dr * wr - di * wi; im[i1] = dr * wi + di * wr;
    __syncthreads();
  }
  // spectral diag filter (ortho scale applied here)
  const float rd = r[ch * HID + ch], id = ii[ch * HID + ch];
  const float rbv = rb[ch], ibv = ib[ch];
#pragma unroll
  for (int e = tid; e < SEQ; e += 256) {
    const float hr = re[e] * INV_SQRT512, hi = im[e] * INV_SQRT512;
    re[e] = fmaxf(hr * rd - hi * id + rbv, 0.f);
    im[e] = fmaxf(hi * rd + hr * id + ibv, 0.f);
  }
  __syncthreads();
  // inverse DIT (e^{+i})
  for (int len = 2; len <= SEQ; len <<= 1) {
    const int half = len >> 1;
    const int grp = tid / half, pos = tid - grp * half;
    const int i0 = grp * len + pos, i1 = i0 + half;
    const float ang = 6.28318530717958647692f * (float)pos / (float)len;
    const float wr = __cosf(ang), wi = __sinf(ang);
    const float vr = re[i1] * wr - im[i1] * wi;
    const float vi = re[i1] * wi + im[i1] * wr;
    const float ur = re[i0], ui = im[i0];
    re[i0] = ur + vr; im[i0] = ui + vi;
    re[i1] = ur - vr; im[i1] = ui - vi;
    __syncthreads();
  }
  const float o0 = re[tid] * INV_SQRT512;
  const float o1 = re[tid + 256] * INV_SQRT512;
  base[(size_t)tid * HID] = o0;
  base[(size_t)(tid + 256) * HID] = o1;
  f16* b16 = h16 + (size_t)b * SEQ * HID + ch;
  b16[(size_t)tid * HID] = (f16)o0;
  b16[(size_t)(tid + 256) * HID] = (f16)o1;
}

__global__ void final_kernel(const float* __restrict__ x2, const float* __restrict__ f2,
                             const float* __restrict__ g, const float* __restrict__ b,
                             float* __restrict__ out) {
  const int idx = blockIdx.x * blockDim.x + threadIdx.x;
  if (idx >= TOK * DIM) return;
  const int ch = idx & 63;
  out[idx] = x2[idx] + f2[idx] * BN_INV_C * g[ch] + b[ch];
}

// ---------------------------------------------------------------------------
// host orchestration
// ---------------------------------------------------------------------------
static inline void* carve(char*& p, size_t bytes) {
  void* r = (void*)p;
  p += (bytes + 255) & ~(size_t)255;
  return r;
}

extern "C" void kernel_launch(void* const* d_in, const int* in_sizes, int n_in,
                              void* d_out, int out_size, void* d_ws, size_t ws_size,
                              hipStream_t stream) {
  (void)in_sizes; (void)n_in; (void)out_size; (void)ws_size;
  // inputs (setup_inputs dict order, params flattened in insertion order)
  const float* x   = (const float*)d_in[0];
  const float* cs1 = (const float*)d_in[1];
  const float* cs2 = (const float*)d_in[2];
  const float* cs3 = (const float*)d_in[3];
  const float* cs4 = (const float*)d_in[4];
  const float* ln_g   = (const float*)d_in[5];
  const float* ln_b   = (const float*)d_in[6];
  const float* in_w   = (const float*)d_in[7];
  const float* conv_w = (const float*)d_in[8];
  const float* conv_b = (const float*)d_in[9];
  const float* xp_w   = (const float*)d_in[10];
  const float* dt_w   = (const float*)d_in[11];
  const float* dt_b   = (const float*)d_in[12];
  const float* A_log  = (const float*)d_in[13];
  const float* Dp     = (const float*)d_in[14];
  const float* cp_w   = (const float*)d_in[15];
  const float* op_w   = (const float*)d_in[16];
  const float* fc1_w  = (const float*)d_in[17];
  const float* bn1_g  = (const float*)d_in[18];
  const float* bn1_b  = (const float*)d_in[19];
  const float* r_m    = (const float*)d_in[20];
  const float* i_m    = (const float*)d_in[21];
  const float* rb_v   = (const float*)d_in[22];
  const float* ib_v   = (const float*)d_in[23];
  const float* fc2_w  = (const float*)d_in[24];
  const float* bn2_g  = (const float*)d_in[25];
  const float* bn2_b  = (const float*)d_in[26];
  const float* n1_g   = (const float*)d_in[27];
  const float* n1_b   = (const float*)d_in[28];
  const float* n2_g   = (const float*)d_in[29];
  const float* n2_b   = (const float*)d_in[30];

  char* wp = (char*)d_ws;
  f16*   wIn   = (f16*)  carve(wp, 256 * 64 * sizeof(f16));
  f16*   wXp   = (f16*)  carve(wp, 128 * 128 * sizeof(f16));
  f16*   wCp   = (f16*)  carve(wp, 64 * 64 * sizeof(f16));
  f16*   wOp   = (f16*)  carve(wp, 64 * 128 * sizeof(f16));
  f16*   wF1   = (f16*)  carve(wp, 128 * 64 * sizeof(f16));
  f16*   wF2   = (f16*)  carve(wp, 64 * 128 * sizeof(f16));
  float* Aneg  = (float*)carve(wp, 128 * 64 * sizeof(float));
  f16*   xn16  = (f16*)  carve(wp, (size_t)TOK * DIM * sizeof(f16));
  float* uz    = (float*)carve(wp, (size_t)TOK * 256 * sizeof(float));
  float* us    = (float*)carve(wp, (size_t)TOK * DI * sizeof(float));
  f16*   us16  = (f16*)  carve(wp, (size_t)TOK * DI * sizeof(f16));
  float* dbl   = (float*)carve(wp, (size_t)TOK * 128 * sizeof(float));
  float* dtbuf = (float*)carve(wp, (size_t)TOK * DI * sizeof(float));
  f16*   c16   = (f16*)  carve(wp, (size_t)TOK * DIM * sizeof(f16));
  float* Cm    = (float*)carve(wp, (size_t)TOK * DIM * sizeof(float));
  float* yv    = (float*)carve(wp, (size_t)TOK * DI * sizeof(float));
  f16*   yg    = (f16*)  carve(wp, (size_t)TOK * DI * sizeof(f16));
  float* yo    = (float*)carve(wp, (size_t)TOK * DIM * sizeof(float));
  float* xo    = (float*)carve(wp, (size_t)TOK * DIM * sizeof(float));
  float* x2    = (float*)carve(wp, (size_t)TOK * DIM * sizeof(float));
  f16*   x2n16 = (f16*)  carve(wp, (size_t)TOK * DIM * sizeof(f16));
  float* hb    = (float*)carve(wp, (size_t)TOK * HID * sizeof(float));
  f16*   h16   = (f16*)  carve(wp, (size_t)TOK * HID * sizeof(f16));
  float* f2b   = (float*)carve(wp, (size_t)TOK * DIM * sizeof(float));

  prep_kernel<<<64, 256, 0, stream>>>(in_w, xp_w, cp_w, op_w, fc1_w, fc2_w, A_log,
                                      wIn, wXp, wCp, wOp, wF1, wF2, Aneg);

  const dim3 gblk(32, 4);

  auto run_ca = [&](const float* cptr, int T, int beta) {
    const int Beff = TOK / T;
    ln_gather_f16_kernel<<<TOK / 8, 256, 0, stream>>>(x, ln_g, ln_b, xn16, T);
    gemm_wmma_kernel<<<dim3(TOK / 64, 256 / 64), gblk, 0, stream>>>(xn16, wIn, uz, 64, 256);
    conv_silu_kernel<<<(TOK * DI) / 256, 256, 0, stream>>>(uz, conv_w, conv_b, us, us16, T);
    gemm_wmma_kernel<<<dim3(TOK / 64, 128 / 64), gblk, 0, stream>>>(us16, wXp, dbl, 128, 128);
    dt_softplus_kernel<<<(TOK * DI) / 256, 256, 0, stream>>>(dbl, dt_w, dt_b, dtbuf);
    cast_f16_kernel<<<(TOK * DIM) / 256, 256, 0, stream>>>(cptr, c16, TOK * DIM);
    gemm_wmma_kernel<<<dim3(TOK / 64, 64 / 64), gblk, 0, stream>>>(c16, wCp, Cm, 64, 64);
    scan_kernel<<<Beff, 128, 0, stream>>>(us, dtbuf, Aneg, dbl, Cm, Dp, yv, T);
    gate_kernel<<<(TOK * DI) / 256, 256, 0, stream>>>(yv, uz, yg);
    gemm_wmma_kernel<<<dim3(TOK / 64, 64 / 64), gblk, 0, stream>>>(yg, wOp, yo, 128, 64);
    scatter_add_kernel<<<(TOK * DIM) / 256, 256, 0, stream>>>(yo, xo, T, beta);
  };

  run_ca(cs1, 512, 0);   // full scale
  run_ca(cs2, 256, 1);   // s = 2
  run_ca(cs3, 128, 1);   // s = 4
  run_ca(cs4, 64, 1);    // s = 8

  ln_add_kernel<<<TOK / 8, 256, 0, stream>>>(xo, x, n1_g, n1_b, x2);
  ln_gather_f16_kernel<<<TOK / 8, 256, 0, stream>>>(x2, n2_g, n2_b, x2n16, 512); // identity perm
  gemm_wmma_kernel<<<dim3(TOK / 64, 128 / 64), gblk, 0, stream>>>(x2n16, wF1, hb, 64, 128);
  bn_relu_kernel<<<(TOK * HID) / 256, 256, 0, stream>>>(hb, bn1_g, bn1_b);
  fft_filter_kernel<<<dim3(8, 128), 256, 0, stream>>>(hb, h16, r_m, i_m, rb_v, ib_v);
  gemm_wmma_kernel<<<dim3(TOK / 64, 64 / 64), gblk, 0, stream>>>(h16, wF2, f2b, 128, 64);
  final_kernel<<<(TOK * DIM) / 256, 256, 0, stream>>>(x2, f2b, bn2_g, bn2_b, (float*)d_out);
}
